// MacCell_51883204935942
// MI455X (gfx1250) — matrix operand order repaired and checked by reading, not compiled, not run on hardware
//
#include <hip/hip_runtime.h>

typedef __attribute__((ext_vector_type(16))) __bf16 v16bf;
typedef __attribute__((ext_vector_type(8)))  float  v8f;

struct U8 { unsigned int x[8]; };

__device__ __forceinline__ unsigned short f2bf(float f) {
  unsigned int u = __float_as_uint(f);
  unsigned int r = u + 0x7FFFu + ((u >> 16) & 1u);
  return (unsigned short)(r >> 16);
}

__device__ __forceinline__ unsigned lds_offset(const void* p) {
  return (unsigned)(unsigned long long)p;   // low 32 bits of flat LDS addr = LDS byte offset
}

#define MODE_F32        0
#define MODE_BF16       1
#define MODE_SCALE_BF16 2
#define MODE_GATE       3

// ---------------------------------------------------------------------------
// f32 -> bf16 convert
// ---------------------------------------------------------------------------
__global__ __launch_bounds__(256)
void f2bf_kernel(const float* __restrict__ in, unsigned short* __restrict__ out, size_t n) {
  size_t i = (size_t)blockIdx.x * 256 + threadIdx.x;
  if (i < n) out[i] = f2bf(in[i]);
}

// ---------------------------------------------------------------------------
// Generic bf16 WMMA GEMM, N fixed = 512.
//   C[row, col] = sum_k A[row,k] * W[k,col]  (+ bias)  (+ epilogue)
// A split: k <  K0A -> A0[row*512 + k], else A1[row*512 + (k-K0A)]
// W split: k <  K0W -> W0[k*512 + col],  else W1[(k-K0W)*512 + col]
// Block tile 128(M) x 64(N), 8 waves, wave = 16 rows x 64 cols
// (4 x v_wmma_f32_16x16x32_bf16 per 32-wide K chunk).
// A tiles are staged with GLOBAL_LOAD_ASYNC_TO_LDS_B128 (ASYNCcnt) into a
// double-buffered LDS pipeline: loads for chunk i+1 overlap WMMA on chunk i.
// ---------------------------------------------------------------------------
template <int MODE>
__global__ __launch_bounds__(256)
void gemm_bf16_n512(const unsigned short* __restrict__ A0,
                    const unsigned short* __restrict__ A1, int K0A,
                    const unsigned short* __restrict__ W0,
                    const unsigned short* __restrict__ W1, int K0W,
                    const float* __restrict__ bias0,
                    const float* __restrict__ bias1,
                    int Ktot,
                    const float* __restrict__ scalev, int rowsPerB,
                    const float* __restrict__ gatev, const float* __restrict__ mi1f,
                    void* __restrict__ outp)
{
  __shared__ __align__(16) unsigned short sA[2][128 * 32];   // [m][k]
  __shared__ __align__(16) unsigned short sW[2][64 * 32];    // [n][k] (transposed)

  const int tid  = threadIdx.x;
  const int lane = tid & 31;
  const int wv   = tid >> 5;
  const int colBase = blockIdx.x * 64;
  const int rowBase = blockIdx.y * 128;

  // ---- tile loader: A via async global->LDS, W via sync load + transpose ----
  auto loadTile = [&](int kc, int buf) {
    // uniform per-chunk piece selection (K0A/K0W are multiples of 32)
    const unsigned short* Ap = (kc < K0A) ? A0 : A1;
    const int kabase         = (kc < K0A) ? kc : kc - K0A;
    const unsigned short* Wp = (kc < K0W) ? W0 : W1;
    const int kwbase         = (kc < K0W) ? kc : kc - K0W;

    // A tile 128x32 bf16: 2 x 16B async copies per thread, direct to LDS
    #pragma unroll
    for (int i = 0; i < 2; ++i) {
      const int c  = tid + i * 256;     // chunk of 8 elems
      const int r  = c >> 2;
      const int co = (c & 3) * 8;
      const unsigned goff = (unsigned)(((rowBase + r) * 512 + kabase + co) * 2);
      const unsigned dlds = lds_offset(&sA[buf][r * 32 + co]);
      asm volatile("global_load_async_to_lds_b128 %0, %1, %2"
                   :: "v"(dlds), "v"(goff), "s"(Ap) : "memory");
    }
    // W tile 32k x 64n bf16: sync load, store transposed [n][k]
    {
      const int k  = tid >> 3;
      const int no = (tid & 7) * 8;
      const uint4 v = *reinterpret_cast<const uint4*>(
          &Wp[(size_t)(kwbase + k) * 512 + colBase + no]);
      const unsigned short* e = reinterpret_cast<const unsigned short*>(&v);
      #pragma unroll
      for (int j = 0; j < 8; ++j) sW[buf][(no + j) * 32 + k] = e[j];
    }
  };

  v8f acc[4];
  #pragma unroll
  for (int i = 0; i < 4; ++i)
    #pragma unroll
    for (int j = 0; j < 8; ++j) acc[i][j] = 0.0f;

  loadTile(0, 0);
  asm volatile("s_wait_asynccnt 0x0" ::: "memory");
  __syncthreads();

  int cur = 0;
  for (int kc = 0; kc < Ktot; kc += 32) {
    if (kc + 32 < Ktot) loadTile(kc + 32, cur ^ 1);

    // ---- A fragment (contiguous k = 16*(lane/16) + s, same order as B) ----
    const int m    = (wv << 4) + (lane & 15);
    const int koff = (lane >> 4) << 4;                // 0 or 16
    U8 ua;
    const unsigned short* ap = &sA[cur][m * 32 + koff];
    #pragma unroll
    for (int j = 0; j < 8; ++j) ua.x[j]     = *reinterpret_cast<const unsigned int*>(&ap[2 * j]);
    const v16bf fa = __builtin_bit_cast(v16bf, ua);

    #pragma unroll
    for (int nt = 0; nt < 4; ++nt) {
      const int n  = (nt << 4) + (lane & 15);
      const int kb = (lane >> 4) << 4;                // 0 or 16
      U8 ub;
      const unsigned short* wp = &sW[cur][n * 32 + kb];
      #pragma unroll
      for (int j = 0; j < 8; ++j) ub.x[j] = *reinterpret_cast<const unsigned int*>(&wp[2 * j]);
      const v16bf fb = __builtin_bit_cast(v16bf, ub);
      acc[nt] = __builtin_amdgcn_wmma_f32_16x16x32_bf16(
          false, fa, false, fb, (short)0, acc[nt], false, false);
    }

    asm volatile("s_wait_asynccnt 0x0" ::: "memory");
    __syncthreads();
    cur ^= 1;
  }

  // ---- epilogue (C/D layout: VGPR r -> M = r + 8*(lane>=16), N = lane&15) ----
  const int rBaseM = rowBase + (wv << 4) + ((lane >> 4) << 3);
  const int nlo    = lane & 15;
  #pragma unroll
  for (int nt = 0; nt < 4; ++nt) {
    const int col = colBase + (nt << 4) + nlo;
    float bb = 0.0f;
    if (bias0) bb += bias0[col];
    if (bias1) bb += bias1[col];
    #pragma unroll
    for (int r = 0; r < 8; ++r) {
      const int row = rBaseM + r;
      const float v = acc[nt][r] + bb;
      const size_t oidx = (size_t)row * 512 + col;
      if constexpr (MODE == MODE_F32) {
        reinterpret_cast<float*>(outp)[oidx] = v;
      } else if constexpr (MODE == MODE_BF16) {
        reinterpret_cast<unsigned short*>(outp)[oidx] = f2bf(v);
      } else if constexpr (MODE == MODE_SCALE_BF16) {
        const int bidx = row / rowsPerB;
        const float s  = scalev[(size_t)bidx * 512 + col];
        reinterpret_cast<unsigned short*>(outp)[oidx] = f2bf(v * s);
      } else { // MODE_GATE
        const float g = gatev[row];
        reinterpret_cast<float*>(outp)[oidx] = g * mi1f[oidx] + (1.0f - g) * v;
      }
    }
  }
}

// ---------------------------------------------------------------------------
// Control stage: cai = dot(cws[b,s,:], cqi[b,:]*W_cai) + b_cai; softmax over S;
// ci[b,:] = sum_s p[s]*cws[b,s,:].   One block per b.
// ---------------------------------------------------------------------------
__global__ __launch_bounds__(256)
void cai_ci_kernel(const float* __restrict__ cqi,
                   const float* __restrict__ cws,
                   const float* __restrict__ Wcai, const float* __restrict__ bcai,
                   float* __restrict__ ci_out)
{
  const int b = blockIdx.x, tid = threadIdx.x, lane = tid & 31, wv = tid >> 5;
  __shared__ float uL[512];
  __shared__ float sL[128];
  __shared__ float invs;

  uL[tid]       = cqi[(size_t)b * 512 + tid]       * Wcai[tid];
  uL[tid + 256] = cqi[(size_t)b * 512 + tid + 256] * Wcai[tid + 256];
  __syncthreads();

  for (int si = 0; si < 16; ++si) {
    const int s = wv + si * 8;
    const size_t rb = ((size_t)b * 128 + s) * 512;
    float p = 0.0f;
    for (int j = 0; j < 16; ++j) { const int c = j * 32 + lane; p += cws[rb + c] * uL[c]; }
    for (int off = 16; off; off >>= 1) p += __shfl_xor(p, off, 32);
    if (lane == 0) sL[s] = p + bcai[0];
  }
  __syncthreads();

  if (tid == 0) {
    float mx = -3.0e38f;
    for (int s = 0; s < 128; ++s) mx = fmaxf(mx, sL[s]);
    float sum = 0.0f;
    for (int s = 0; s < 128; ++s) { const float e = __expf(sL[s] - mx); sL[s] = e; sum += e; }
    invs = 1.0f / sum;
  }
  __syncthreads();

  float a0 = 0.0f, a1 = 0.0f;
  for (int s = 0; s < 128; ++s) {
    const float e = sL[s];
    const size_t rb = ((size_t)b * 128 + s) * 512;
    a0 += e * cws[rb + tid];
    a1 += e * cws[rb + tid + 256];
  }
  const float inv = invs;
  ci_out[(size_t)b * 512 + tid]       = a0 * inv;
  ci_out[(size_t)b * 512 + tid + 256] = a1 * inv;
}

// ---------------------------------------------------------------------------
// Double softmax over 14x14 (axis H, then axis W of the probs) + ri reduction.
// One thread per (b,d); rai (103MB) is L2-resident on MI455X (192MB L2).
// ---------------------------------------------------------------------------
__global__ __launch_bounds__(256)
void rai_softmax_ri_kernel(const float* __restrict__ rai,
                           const float* __restrict__ Kt,
                           unsigned short* __restrict__ ri_bf)
{
  const int gid = blockIdx.x * 256 + threadIdx.x;      // 131072 = 256*512
  const int b = gid >> 9;
  const int d = gid & 511;
  const size_t base = (size_t)b * 196 * 512 + d;

  float maxh[14], sumh[14];
  #pragma unroll
  for (int w = 0; w < 14; ++w) { maxh[w] = -3.0e38f; sumh[w] = 0.0f; }

  // pass 1: online softmax stats over H for each w
  for (int h = 0; h < 14; ++h) {
    #pragma unroll
    for (int w = 0; w < 14; ++w) {
      const float x = rai[base + (size_t)(h * 14 + w) * 512];
      const float m = fmaxf(maxh[w], x);
      sumh[w] = sumh[w] * __expf(maxh[w] - m) + __expf(x - m);
      maxh[w] = m;
    }
  }
  float rsumh[14];
  #pragma unroll
  for (int w = 0; w < 14; ++w) rsumh[w] = 1.0f / sumh[w];

  // pass 2: per h row: second softmax over W of p1, accumulate ri
  float acc = 0.0f;
  for (int h = 0; h < 14; ++h) {
    float p1v[14];
    float mh = -3.0e38f;
    #pragma unroll
    for (int w = 0; w < 14; ++w) {
      const float x = rai[base + (size_t)(h * 14 + w) * 512];
      const float p1 = __expf(x - maxh[w]) * rsumh[w];
      p1v[w] = p1;
      mh = fmaxf(mh, p1);
    }
    float sw = 0.0f;
    #pragma unroll
    for (int w = 0; w < 14; ++w) { const float e = __expf(p1v[w] - mh); p1v[w] = e; sw += e; }
    const float rsw = 1.0f / sw;
    #pragma unroll
    for (int w = 0; w < 14; ++w)
      acc += (p1v[w] * rsw) * Kt[base + (size_t)(h * 14 + w) * 512];
  }
  ri_bf[gid] = f2bf(acc);
}

// ---------------------------------------------------------------------------
// Write stage (per b): saij = softmax_T(dot(ci*C_past[t], W_wcc)+b_wcc),
// mi_sa = sum_t saij*M_past[t]; gate = sigmoid(dot(ci,W_wci)+b_wci).
// ---------------------------------------------------------------------------
__global__ __launch_bounds__(256)
void write_stage_kernel(const float* __restrict__ ci,
                        const float* __restrict__ C_past,
                        const float* __restrict__ M_past,
                        const float* __restrict__ Wwcc, const float* __restrict__ bwcc,
                        const float* __restrict__ Wwci, const float* __restrict__ bwci,
                        unsigned short* __restrict__ mi_sa_bf,
                        float* __restrict__ gate)
{
  const int b = blockIdx.x, tid = threadIdx.x, lane = tid & 31, wv = tid >> 5;
  __shared__ float ciL[512];
  __shared__ float red[256];
  __shared__ float sL[16];

  ciL[tid]       = ci[(size_t)b * 512 + tid];
  ciL[tid + 256] = ci[(size_t)b * 512 + tid + 256];
  __syncthreads();

  for (int t = wv; t < 12; t += 8) {
    const size_t cb = ((size_t)b * 12 + t) * 512;
    float p = 0.0f;
    for (int j = 0; j < 16; ++j) { const int c = j * 32 + lane; p += ciL[c] * C_past[cb + c] * Wwcc[c]; }
    for (int off = 16; off; off >>= 1) p += __shfl_xor(p, off, 32);
    if (lane == 0) sL[t] = p + bwcc[0];
  }
  red[tid] = ciL[tid] * Wwci[tid] + ciL[tid + 256] * Wwci[tid + 256];
  __syncthreads();

  if (tid == 0) {
    float g = 0.0f;
    for (int i = 0; i < 256; ++i) g += red[i];
    g += bwci[0];
    gate[b] = 1.0f / (1.0f + __expf(-g));
    float mx = -3.0e38f;
    for (int t = 0; t < 12; ++t) mx = fmaxf(mx, sL[t]);
    float s = 0.0f;
    for (int t = 0; t < 12; ++t) { const float e = __expf(sL[t] - mx); sL[t] = e; s += e; }
    const float inv = 1.0f / s;
    for (int t = 0; t < 12; ++t) sL[t] *= inv;
  }
  __syncthreads();

  float a0 = 0.0f, a1 = 0.0f;
  for (int t = 0; t < 12; ++t) {
    const float p = sL[t];
    const size_t mb = ((size_t)b * 12 + t) * 512;
    a0 += p * M_past[mb + tid];
    a1 += p * M_past[mb + tid + 256];
  }
  mi_sa_bf[(size_t)b * 512 + tid]       = f2bf(a0);
  mi_sa_bf[(size_t)b * 512 + tid + 256] = f2bf(a1);
}

// ---------------------------------------------------------------------------
// Host orchestration
// ---------------------------------------------------------------------------
extern "C" void kernel_launch(void* const* d_in, const int* in_sizes, int n_in,
                              void* d_out, int out_size, void* d_ws, size_t ws_size,
                              hipStream_t stream) {
  (void)in_sizes; (void)n_in; (void)out_size; (void)ws_size;

  const float* ci_1   = (const float*)d_in[0];
  const float* mi_1   = (const float*)d_in[1];
  const float* q      = (const float*)d_in[2];
  const float* cws    = (const float*)d_in[3];
  const float* Kin    = (const float*)d_in[4];
  const float* C_past = (const float*)d_in[5];
  const float* M_past = (const float*)d_in[6];
  const float* W_cq   = (const float*)d_in[7];  const float* b_cq   = (const float*)d_in[8];
  const float* W_cqi  = (const float*)d_in[9];  const float* b_cqi  = (const float*)d_in[10];
  const float* W_cai  = (const float*)d_in[11]; const float* b_cai  = (const float*)d_in[12];
  const float* W_rm   = (const float*)d_in[13]; const float* b_rm   = (const float*)d_in[14];
  const float* W_rk   = (const float*)d_in[15]; const float* b_rk   = (const float*)d_in[16];
  const float* W_rik  = (const float*)d_in[17]; const float* b_rik  = (const float*)d_in[18];
  const float* W_rci  = (const float*)d_in[19]; const float* b_rci  = (const float*)d_in[20];
  const float* W_wrm  = (const float*)d_in[21]; const float* b_wrm  = (const float*)d_in[22];
  const float* W_wcc  = (const float*)d_in[23]; const float* b_wcc  = (const float*)d_in[24];
  const float* W_wsa  = (const float*)d_in[25]; const float* b_wsa  = (const float*)d_in[26];
  const float* W_winfo= (const float*)d_in[27]; const float* b_winfo= (const float*)d_in[28];
  const float* W_wci  = (const float*)d_in[29]; const float* b_wci  = (const float*)d_in[30];

  const size_t NB = 50176;        // B*H*W
  const size_t Dm = 512, Bm = 256;
  const size_t SM = Bm * Dm;      // 131072

  char* ws = (char*)d_ws;
  size_t off = 0;
  auto take = [&](size_t bytes) -> char* {
    char* p = ws + off;
    off = (off + bytes + 255) & ~(size_t)255;
    return p;
  };

  // region A: Kbf + Iibf contiguous; later reused as rai (f32, same byte size)
  unsigned short* Kbf  = (unsigned short*)take(NB * Dm * 2);
  unsigned short* Iibf = (unsigned short*)take(NB * Dm * 2);
  float*          raiF = (float*)Kbf;
  unsigned short* Xbf  = (unsigned short*)take(NB * Dm * 2);

  unsigned short* bWcq   = (unsigned short*)take(512 * 512 * 2);
  unsigned short* bWcqi  = (unsigned short*)take(1024 * 512 * 2);
  unsigned short* bWrm   = (unsigned short*)take(512 * 512 * 2);
  unsigned short* bWrk   = (unsigned short*)take(512 * 512 * 2);
  unsigned short* bWrik  = (unsigned short*)take(1024 * 512 * 2);
  unsigned short* bWrci  = (unsigned short*)take(512 * 512 * 2);
  unsigned short* bWwrm  = (unsigned short*)take(1024 * 512 * 2);
  unsigned short* bWwsa  = (unsigned short*)take(512 * 512 * 2);
  unsigned short* bWwinfo= (unsigned short*)take(512 * 512 * 2);

  unsigned short* qbf     = (unsigned short*)take(SM * 2);
  unsigned short* ci1bf   = (unsigned short*)take(SM * 2);
  unsigned short* mi1bf   = (unsigned short*)take(SM * 2);
  unsigned short* qibf    = (unsigned short*)take(SM * 2);
  unsigned short* ribf    = (unsigned short*)take(SM * 2);
  unsigned short* misabf  = (unsigned short*)take(SM * 2);
  unsigned short* miinfobf= (unsigned short*)take(SM * 2);

  float* rmF   = (float*)take(SM * 4);
  float* cqiF  = (float*)take(SM * 4);
  float* gateF = (float*)take(Bm * 4);

  float* out_ci = (float*)d_out;
  float* out_mi = out_ci + SM;

  auto conv = [&](const float* src, unsigned short* dst, size_t n) {
    const int blocks = (int)((n + 255) / 256);
    f2bf_kernel<<<blocks, 256, 0, stream>>>(src, dst, n);
  };

  // ---- converts ----
  conv(q, qbf, SM);  conv(ci_1, ci1bf, SM);  conv(mi_1, mi1bf, SM);
  conv(Kin, Kbf, NB * Dm);
  conv(W_cq, bWcq, 512*512);     conv(W_cqi, bWcqi, 1024*512);
  conv(W_rm, bWrm, 512*512);     conv(W_rk, bWrk, 512*512);
  conv(W_rik, bWrik, 1024*512);  conv(W_rci, bWrci, 512*512);
  conv(W_wrm, bWwrm, 1024*512);  conv(W_wsa, bWwsa, 512*512);
  conv(W_winfo, bWwinfo, 512*512);

  const dim3 blk(256);
  const dim3 gSmall(8, 2);     // M=256
  const dim3 gBig(8, 392);     // M=50176

  // rm = mi_1 @ W_rm + b_rm   (f32)
  gemm_bf16_n512<MODE_F32><<<gSmall, blk, 0, stream>>>(mi1bf, nullptr, 512, bWrm, nullptr, 512,
      b_rm, nullptr, 512, nullptr, 1, nullptr, nullptr, rmF);
  // qi = q @ W_cq + b_cq   (bf16)
  gemm_bf16_n512<MODE_BF16><<<gSmall, blk, 0, stream>>>(qbf, nullptr, 512, bWcq, nullptr, 512,
      b_cq, nullptr, 512, nullptr, 1, nullptr, nullptr, qibf);
  // cqi = [ci_1 | qi] @ W_cqi + b_cqi   (f32)
  gemm_bf16_n512<MODE_F32><<<gSmall, blk, 0, stream>>>(ci1bf, qibf, 512, bWcqi, nullptr, 1024,
      b_cqi, nullptr, 1024, nullptr, 1, nullptr, nullptr, cqiF);
  // cai softmax + ci
  cai_ci_kernel<<<256, 256, 0, stream>>>(cqiF, cws, W_cai, b_cai, out_ci);

  // Ii = rm * (K @ W_rk + b_rk)   (bf16)
  gemm_bf16_n512<MODE_SCALE_BF16><<<gBig, blk, 0, stream>>>(Kbf, nullptr, 512, bWrk, nullptr, 512,
      b_rk, nullptr, 512, rmF, 196, nullptr, nullptr, Iibf);
  // X = ci * ([Ii | K] @ W_rik + b_rik)   (bf16)
  gemm_bf16_n512<MODE_SCALE_BF16><<<gBig, blk, 0, stream>>>(Iibf, Kbf, 512, bWrik, nullptr, 1024,
      b_rik, nullptr, 1024, out_ci, 196, nullptr, nullptr, Xbf);
  // rai = X @ W_rci + b_rci   (f32, overwrites Kbf/Iibf region — both now dead)
  gemm_bf16_n512<MODE_F32><<<gBig, blk, 0, stream>>>(Xbf, nullptr, 512, bWrci, nullptr, 512,
      b_rci, nullptr, 512, nullptr, 1, nullptr, nullptr, raiF);
  // double softmax + ri
  rai_softmax_ri_kernel<<<512, 256, 0, stream>>>(raiF, Kin, ribf);

  // mi_info = [ri | mi_1] @ W_wrm + b_wrm   (bf16)
  gemm_bf16_n512<MODE_BF16><<<gSmall, blk, 0, stream>>>(ribf, mi1bf, 512, bWwrm, nullptr, 1024,
      b_wrm, nullptr, 1024, nullptr, 1, nullptr, nullptr, miinfobf);
  // saij softmax, mi_sa, gate
  write_stage_kernel<<<256, 256, 0, stream>>>(out_ci, C_past, M_past,
      W_wcc, b_wcc, W_wci, b_wci, misabf, gateF);
  // mi = gate*mi_1 + (1-gate)*([mi_sa|mi_info] @ [W_wsa;W_winfo] + b_wsa + b_winfo)
  gemm_bf16_n512<MODE_GATE><<<gSmall, blk, 0, stream>>>(misabf, miinfobf, 512, bWwsa, bWwinfo, 512,
      b_wsa, b_winfo, 1024, nullptr, 1, gateF, mi_1, out_mi);
}
